// leastActionNet_309237645489
// MI455X (gfx1250) — matrix operand order repaired and checked by reading, not compiled, not run on hardware
//
#include <hip/hip_runtime.h>

typedef __attribute__((ext_vector_type(16))) _Float16 v16h;
typedef __attribute__((ext_vector_type(8)))  float    v8f;

#define NLAYERS 8
#define NCH 64
#define HH 96
#define WW 96
#define BB 8
constexpr size_t PLANE = (size_t)HH * WW;                  // 9216
constexpr size_t NP    = (size_t)BB * NCH * PLANE;          // 4,718,592
constexpr size_t ROWCL = (size_t)WW * NCH;                  // 6144 elems per (b,y) row, channel-last
constexpr int    NLB   = NLAYERS * BB;                      // 64

// ---------------------------------------------------------------- utilities
__global__ void zero_f32(float* p, size_t n) {
  size_t i = (size_t)blockIdx.x * blockDim.x + threadIdx.x;
  if (i < n) p[i] = 0.f;
}

// Convert weights to f16 in [l][tap][co][ci] order.
// Wf: forward conv   -> Wf[l][kh*3+kw][co][ci] = K[l][co][ci][kh][kw]
// Wb: conv_transpose -> Wb[l][kh*3+kw][o][i]   = K[l][i][o][2-kh][2-kw]  (tap' = 8-tap)
__global__ void prep_weights(const float* __restrict__ K,
                             _Float16* __restrict__ Wf, _Float16* __restrict__ Wb) {
  int t = blockIdx.x * blockDim.x + threadIdx.x;
  if (t >= NLAYERS * 9 * NCH * NCH) return;
  int l   = t / (9 * NCH * NCH);
  int r   = t % (9 * NCH * NCH);
  int tap = r / (NCH * NCH);
  int r2  = r % (NCH * NCH);
  int o   = r2 / NCH;
  int i   = r2 % NCH;
  size_t dst = ((size_t)(l * 9 + tap) * NCH + o) * NCH + i;
  Wf[dst] = (_Float16)K[(((size_t)l * NCH + o) * NCH + i) * 9 + tap];
  Wb[dst] = (_Float16)K[(((size_t)l * NCH + i) * NCH + o) * 9 + (8 - tap)];
}

// NCHW fp32 -> channel-last fp32 [b][y][x][c]
__global__ void to_cl(const float* __restrict__ X, float* __restrict__ Xcl, int nb) {
  size_t i = (size_t)blockIdx.x * blockDim.x + threadIdx.x;
  if (i >= (size_t)nb * NCH * PLANE) return;
  int c = (int)(i & 63);
  size_t s = i >> 6;
  int x = (int)(s % WW); s /= WW;
  int y = (int)(s % HH);
  int b = (int)(s / HH);
  Xcl[i] = X[((size_t)(b * NCH + c) * HH + y) * WW + x];
}

__global__ void finalize_stats(const float* __restrict__ ssum, const float* __restrict__ ssq,
                               float* __restrict__ ssc, float* __restrict__ sbi) {
  int i = blockIdx.x * blockDim.x + threadIdx.x;
  if (i >= NLB * NCH) return;
  float inv = 1.0f / (float)PLANE;
  float m = ssum[i] * inv;
  float v = ssq[i] * inv - m * m;         // biased variance
  float r = rsqrtf(v + 1e-5f);
  ssc[i] = r;
  sbi[i] = -m * r;
}

// act = f16(lrelu(dZ * scale + bias)), channel-last in/out
__global__ void norm_act(const float* __restrict__ dZ,
                         const float* __restrict__ ssc, const float* __restrict__ sbi,
                         _Float16* __restrict__ act) {
  size_t i = (size_t)blockIdx.x * blockDim.x + threadIdx.x;
  if (i >= (size_t)NLAYERS * NP) return;
  int lb = (int)(i / (PLANE * NCH));
  int c  = (int)(i & 63);
  float t = fmaf(dZ[i], ssc[lb * NCH + c], sbi[lb * NCH + c]);
  t = t >= 0.f ? t : 0.2f * t;
  act[i] = (_Float16)t;
}

// ------------------------------------------------------------ WMMA conv pass
// One block = one (layer l, batch b, output row y); 192 threads = 6 waves.
// Input: fp16 channel-last [lb][y][x][ci]; staged into LDS rows y-1..y+1 via
// CDNA5 async global->LDS DMA (ASYNCcnt). Wave w computes pixel tile
// [16w,16w+15] x all 64 output channels via v_wmma_f32_16x16x32_f16.
// Dir==0: dZ = conv2d(Z)   (channel-last fp32) + per-(b,c) stats atomics.
// Dir==1: Y = -convT(act) (+Xcl at l=7, +X0cl at l=0), channel-last fp32.
template <int Dir>
__global__ void __launch_bounds__(192)
conv_wmma(const _Float16* __restrict__ inh, float* __restrict__ out,
          const _Float16* __restrict__ Wall,
          float* __restrict__ ssum, float* __restrict__ ssq,
          const float* __restrict__ Xcl, const float* __restrict__ X0cl) {
  __shared__ __align__(16) _Float16 lds_in[3 * 98 * NCH];   // [row][xpad][ci]

  const int bid = blockIdx.x;
  const int y  = bid % HH;
  const int b  = (bid / HH) % BB;
  const int l  = bid / (HH * BB);
  const int lb = l * BB + b;
  const int tid  = threadIdx.x;
  const int wave = tid >> 5;
  const int lane = tid & 31;

  // zero x-padding columns (3 rows * 64 ch = 192 == blockDim)
  { int r = tid >> 6, ci = tid & 63;
    lds_in[(r * 98 + 0)  * NCH + ci] = (_Float16)0.f;
    lds_in[(r * 98 + 97) * NCH + ci] = (_Float16)0.f; }

  // ---- async DMA staging of the 3 input rows (2 waves per row, 12x16B per lane)
  {
    const int r   = wave >> 1;           // 0..2
    const int sub = wave & 1;
    const int yy  = y + r - 1;
    const unsigned ldsRow = (unsigned)((r * 98 + 1) * NCH * 2);  // byte off of [r][x=1][0]
    unsigned lbase;
    {
      auto lp = (__attribute__((address_space(3))) _Float16*)lds_in;
      lbase = (unsigned)(unsigned long long)lp + ldsRow;
    }
    if (yy >= 0 && yy < HH) {
      const unsigned long long gbase = (unsigned long long)(size_t)inh;
      const unsigned grow = (unsigned)((lb * HH + yy) * (int)ROWCL * 2);
#pragma unroll
      for (int i = 0; i < 12; ++i) {
        unsigned ch = (unsigned)((sub * 384 + i * 32 + lane) * 16);
        unsigned la = lbase + ch;
        unsigned ga = grow + ch;
        asm volatile("global_load_async_to_lds_b128 %0, %1, %2"
                     :: "v"(la), "v"(ga), "s"(gbase) : "memory");
      }
    } else {
      uint4 zz; zz.x = 0u; zz.y = 0u; zz.z = 0u; zz.w = 0u;
#pragma unroll
      for (int i = 0; i < 12; ++i) {
        unsigned ch = (unsigned)((sub * 384 + i * 32 + lane) * 16);
        *(uint4*)((char*)lds_in + ldsRow + ch) = zz;
      }
    }
    asm volatile("s_wait_asynccnt 0x0" ::: "memory");
  }
  __syncthreads();

  const int hi = lane >> 4;     // lane half: selects K sub-block per ISA layout
  const int ln = lane & 15;     // M index (A) / N index (B,D)
  const int x0 = wave * 16;

  const _Float16* Wl = Wall + (size_t)l * 9 * NCH * NCH;

  v8f acc[4] = {v8f{}, v8f{}, v8f{}, v8f{}};

  for (int tap = 0; tap < 9; ++tap) {
    const int kh = tap / 3, kw = tap - kh * 3;
    // A: 16x32 f16. lane(ln,hi): halves 0..7 -> K=hi*8+0..7 ; 8..15 -> K=16+hi*8+0..7
    const _Float16* arow = &lds_in[((kh * 98) + x0 + ln + kw) * NCH];
#pragma unroll
    for (int c32 = 0; c32 < 2; ++c32) {
      const int cib = c32 * 32;
      union { v16h v; uint4 q[2]; } A;
      A.q[0] = *(const uint4*)(arow + cib + hi * 8);
      A.q[1] = *(const uint4*)(arow + cib + 16 + hi * 8);
      // B: 32x16 f16. lane n=ln, halves 0..7 -> K=hi*8+0..7 ; 8..15 -> K=16+hi*8+0..7
      const _Float16* wtap = Wl + (size_t)tap * NCH * NCH + cib + hi * 8;
#pragma unroll
      for (int nt = 0; nt < 4; ++nt) {
        union { v16h v; uint4 q[2]; } Bf;
        const uint4* bp = (const uint4*)(wtap + (size_t)(nt * 16 + ln) * NCH);
        Bf.q[0] = bp[0];
        Bf.q[1] = bp[2];
        acc[nt] = __builtin_amdgcn_wmma_f32_16x16x32_f16(
            false, A.v, false, Bf.v, (short)0, acc[nt], false, false);
      }
    }
  }

  // D layout: lane ln = column n (=co), VGPR r -> row m = r + 8*hi (= pixel x0+m)
  // channel-last stores: lane index = channel -> contiguous
  float* outL = out + (size_t)(lb * HH + y) * ROWCL;
  const float* XrowB = Xcl  + (size_t)(b * HH + y) * ROWCL;   // Dir==1, l==7
  const float* Xrow0 = X0cl + (size_t)y * ROWCL;              // Dir==1, l==0
#pragma unroll
  for (int nt = 0; nt < 4; ++nt) {
    const int co = nt * 16 + ln;
    float s = 0.f, q = 0.f;
#pragma unroll
    for (int r = 0; r < 8; ++r) {
      float vv = acc[nt][r];
      int xp = x0 + r + 8 * hi;
      size_t off = (size_t)xp * NCH + co;
      if (Dir == 0) {
        outL[off] = vv;
        s += vv; q += vv * vv;
      } else {
        float o = -vv;
        if (l == NLAYERS - 1) o += XrowB[off];
        if (l == 0)           o += Xrow0[off];
        outL[off] = o;
      }
    }
    if (Dir == 0) {
      s += __shfl_xor(s, 16, 32);
      q += __shfl_xor(q, 16, 32);
      if (hi == 0) {
        atomicAdd(&ssum[lb * NCH + co], s);
        atomicAdd(&ssq[lb * NCH + co], q);
      }
    }
  }
}

// -------------------------------------------------- tridiagonal layer scans
// Y,Z channel-last. Also emits fp16 Z copy (Zh) for the next conv pass.
__global__ void tridiag_kernel(const float* __restrict__ Y, float* __restrict__ Z,
                               _Float16* __restrict__ Zh) {
  size_t p = (size_t)blockIdx.x * blockDim.x + threadIdx.x;
  if (p >= NP) return;
  float yf[NLAYERS];
  float c = 0.f;
#pragma unroll
  for (int i = 0; i < NLAYERS; ++i) {
    float ai = sqrtf((i + 1.0f) / (i + 2.0f));
    float bi = sqrtf((float)i / (i + 1.0f));
    c = ai * (bi * c + Y[(size_t)i * NP + p]);
    yf[i] = c;
  }
  float d = 0.f;
#pragma unroll
  for (int i = NLAYERS - 1; i >= 0; --i) {
    float ai = sqrtf((i + 1.0f) / (i + 2.0f));
    d = ai * (ai * d + yf[i]);
    Z[(size_t)i * NP + p] = d;
    Zh[(size_t)i * NP + p] = (_Float16)d;
  }
}

// channel-last Z -> NCHW out = concat(Z[7], Z[0..7]); LDS tile transpose
__global__ void __launch_bounds__(256) transpose_out(const float* __restrict__ Zf,
                                                     float* __restrict__ out) {
  __shared__ float tile[WW * 65];   // [x][c] with pad to avoid bank conflicts
  int bid = blockIdx.x;
  int y = bid % HH;
  int b = (bid / HH) % BB;
  int l = bid / (HH * BB);
  const float* src = Zf + (size_t)l * NP + (size_t)(b * HH + y) * ROWCL;
  for (int i = threadIdx.x; i < (int)ROWCL; i += 256) {
    int x = i >> 6, c = i & 63;
    tile[x * 65 + c] = src[i];
  }
  __syncthreads();
  for (int j = threadIdx.x; j < (int)ROWCL; j += 256) {
    int c = j / WW, x = j - c * WW;
    float v = tile[x * 65 + c];
    size_t o = ((size_t)(b * NCH + c) * HH + y) * WW + x;
    out[NP + (size_t)l * NP + o] = v;
    if (l == NLAYERS - 1) out[o] = v;
  }
}

// ---------------------------------------------------------------- launcher
extern "C" void kernel_launch(void* const* d_in, const int* in_sizes, int n_in,
                              void* d_out, int out_size, void* d_ws, size_t ws_size,
                              hipStream_t stream) {
  const float* X  = (const float*)d_in[0];
  const float* K  = (const float*)d_in[1];
  const float* X0 = (const float*)d_in[2];
  float* out = (float*)d_out;

  float* Zf   = (float*)d_ws;                       // [8][NP] fp32 channel-last
  float* Dbuf = Zf + (size_t)NLAYERS * NP;          // dZ, then Y (aliased)
  float* Xcl  = Dbuf + (size_t)NLAYERS * NP;        // [NP]
  float* X0cl = Xcl + NP;                           // [NP/8]
  float* ssum = X0cl + NP / BB;
  float* ssq  = ssum + (size_t)NLB * NCH;
  float* ssc  = ssq  + (size_t)NLB * NCH;
  float* sbi  = ssc  + (size_t)NLB * NCH;
  _Float16* Wf = (_Float16*)(sbi + (size_t)NLB * NCH);
  _Float16* Wb = Wf + (size_t)NLAYERS * 9 * NCH * NCH;
  _Float16* Zh = Wb + (size_t)NLAYERS * 9 * NCH * NCH;  // [8][NP] fp16 channel-last
  _Float16* A1 = Zh + (size_t)NLAYERS * NP;             // activations for convT

  const size_t nZ = (size_t)NLAYERS * NP;
  // Zh must start at zero every call (8NP halves == 4NP floats)
  zero_f32<<<(unsigned)((nZ / 2 + 255) / 256), 256, 0, stream>>>((float*)Zh, nZ / 2);
  prep_weights<<<(NLAYERS * 9 * NCH * NCH + 255) / 256, 256, 0, stream>>>(K, Wf, Wb);
  to_cl<<<(unsigned)((NP + 255) / 256), 256, 0, stream>>>(X, Xcl, BB);
  to_cl<<<(unsigned)((NP / BB + 255) / 256), 256, 0, stream>>>(X0, X0cl, 1);

  const int nconv = NLAYERS * BB * HH;   // 6144 blocks
  for (int it = 0; it < 4; ++it) {
    zero_f32<<<(2 * NLB * NCH + 255) / 256, 256, 0, stream>>>(ssum, (size_t)2 * NLB * NCH);
    conv_wmma<0><<<nconv, 192, 0, stream>>>(Zh, Dbuf, Wf, ssum, ssq, nullptr, nullptr);
    finalize_stats<<<(NLB * NCH + 255) / 256, 256, 0, stream>>>(ssum, ssq, ssc, sbi);
    norm_act<<<(unsigned)((nZ + 255) / 256), 256, 0, stream>>>(Dbuf, ssc, sbi, A1);
    conv_wmma<1><<<nconv, 192, 0, stream>>>(A1, Dbuf, Wb, nullptr, nullptr, Xcl, X0cl);
    tridiag_kernel<<<(unsigned)((NP + 255) / 256), 256, 0, stream>>>(Dbuf, Zf, Zh);
  }
  transpose_out<<<nconv, 256, 0, stream>>>(Zf, out);
}
